// Mutil_Path_Fuse_Module_49125835931693
// MI455X (gfx1250) — matrix operand order repaired and checked by reading, not compiled, not run on hardware
//
#include <hip/hip_runtime.h>
#include <hip/hip_bf16.h>

// ---------------------------------------------------------------------------
// MI455X / gfx1250 implementation.
//
// All heavy math on v_wmma_f32_16x16x32_bf16 (fp32 demoted to bf16 operands,
// f32 accumulation):
//   - mask @ x gather-average  -> GEMM (M=512, K=512, N=50176)
//   - all convolutions         -> per-image implicit GEMM
// Per-image class-based branch selection halves the char/text conv work.
// 1x1 convs stage their B-tiles with the Tensor Data Mover
// (tensor_load_to_lds + s_wait_tensorcnt), double-buffered in LDS.
// ---------------------------------------------------------------------------

typedef __attribute__((ext_vector_type(16))) __bf16 v16bf;
typedef __attribute__((ext_vector_type(8)))  float  v8f;
typedef __attribute__((ext_vector_type(4)))  unsigned int v4u;
typedef __attribute__((ext_vector_type(8)))  int v8i;
typedef __attribute__((ext_vector_type(4)))  int v4i;

#if defined(__HIP_DEVICE_COMPILE__) && __has_builtin(__builtin_amdgcn_tensor_load_to_lds)
#define HAVE_TDM 1
#else
#define HAVE_TDM 0
#endif

union ABfrag {
    v16bf v;
    unsigned int u[8];
    unsigned short h[16];
};

__device__ __forceinline__ unsigned short f2bf(float f) {
    union { float f; unsigned int u; } c;
    c.f = f;
    unsigned int u = c.u;
    unsigned int r = u + 0x7FFFu + ((u >> 16) & 1u);   // round-to-nearest-even
    return (unsigned short)(r >> 16);
}

#define FEAT_N 512
#define FEAT_C 256
#define FEAT_HW 196
#define COLS 50176           // C*H*W

// ---------------------------------------------------------------------------
// Weight packing: fp32 [O=256][I=256][khw] -> bf16 [O][K], k = kpos*256+ci.
// ---------------------------------------------------------------------------
__global__ __launch_bounds__(256)
void pack_w(const float* __restrict__ w, unsigned short* __restrict__ out, int khw) {
    int K = khw * 256;
    int idx = blockIdx.x * 256 + threadIdx.x;
    if (idx >= 256 * K) return;
    int o = idx / K;
    int k = idx - o * K;
    int kpos = k >> 8;
    int ci   = k & 255;
    out[idx] = f2bf(w[(size_t)(o * 256 + ci) * khw + kpos]);
}

// Per-channel BN folding: s = gamma*rsqrt(var+eps), b = beta - mean*s.
__global__ __launch_bounds__(256)
void bn_prep(const float* __restrict__ gamma, const float* __restrict__ beta,
             const float* __restrict__ mean,  const float* __restrict__ var,
             float* __restrict__ s, float* __restrict__ b) {
    int c = threadIdx.x;
    float sc = gamma[c] * rsqrtf(var[c] + 1e-5f);
    s[c] = sc;
    b[c] = beta[c] - mean[c] * sc;
}

// ---------------------------------------------------------------------------
// Pairwise box mask (bf16 0/1) + per-row merged scale:
//   selscale[i] = (classes[i]==0 && count>1) ? 1/count : 0
// ---------------------------------------------------------------------------
__global__ __launch_bounds__(512)
void mask_kernel(const float* __restrict__ boxes, const int* __restrict__ classes,
                 unsigned short* __restrict__ maskb, float* __restrict__ selscale) {
    __shared__ float red[512];
    int i = blockIdx.x, j = threadIdx.x;
    float xi0 = boxes[i * 4 + 0], yi0 = boxes[i * 4 + 1];
    float xi1 = boxes[i * 4 + 2], yi1 = boxes[i * 4 + 3];
    float xj0 = boxes[j * 4 + 0], yj0 = boxes[j * 4 + 1];
    float xj1 = boxes[j * 4 + 2], yj1 = boxes[j * 4 + 3];
    float areaj = (xj1 - xj0) * (yj1 - yj0);
    float lx = fmaxf(xi0, xj0), ly = fmaxf(yi0, yj0);
    float rx = fminf(xi1, xj1), ry = fminf(yi1, yj1);
    float w = fmaxf(rx - lx, 0.f), h = fmaxf(ry - ly, 0.f);
    float m = ((w * h) / areaj > 0.9f) ? 1.f : 0.f;
    maskb[(size_t)i * 512 + j] = f2bf(m);
    red[j] = m;
    __syncthreads();
    for (int s = 256; s > 0; s >>= 1) {
        if (j < s) red[j] += red[j + s];
        __syncthreads();
    }
    if (j == 0) {
        float cnt = red[0];
        selscale[i] = (classes[i] == 0 && cnt > 1.0f) ? (1.0f / cnt) : 0.0f;
    }
}

// ---------------------------------------------------------------------------
// Fragment helpers (A/B k order here: contiguous, k = 16*hi + s).
// ---------------------------------------------------------------------------
__device__ __forceinline__ void load_a_frag(ABfrag& a, const unsigned short* ap) {
    // ap = row base + chunk*32 + hi*16 ; pairs at 0,2,4,...,14 (u16 units)
    #pragma unroll
    for (int p = 0; p < 8; ++p) {
        int k0 = p * 2;
        a.u[p] = *(const unsigned int*)(ap + k0);
    }
}

template <int MODE>
__device__ __forceinline__ void conv_epilogue(
        const v8f* acc, int n, int mbase, int wave, int lane16, int hi, int colbase,
        const float* __restrict__ xf, const float* __restrict__ gcf,
        const float* __restrict__ bnS, const float* __restrict__ bnB,
        unsigned short* __restrict__ out_b, float* __restrict__ out_f) {
    #pragma unroll
    for (int s = 0; s < 4; ++s) {
        int pixel = colbase + s * 16 + lane16;
        if (pixel >= FEAT_HW) continue;
        #pragma unroll
        for (int r = 0; r < 8; ++r) {
            int c = mbase + wave * 16 + r + hi * 8;
            size_t idx = ((size_t)n * FEAT_C + c) * FEAT_HW + pixel;
            float v = acc[s][r];
            if constexpr (MODE == 0) {
                out_b[idx] = f2bf(v);
            } else if constexpr (MODE == 1) {
                out_b[idx] = f2bf(v + xf[idx] + gcf[idx]);
            } else {
                out_f[idx] = fmaxf(fmaf(v, bnS[c], bnB[c]), 0.f);
            }
        }
    }
}

// ---------------------------------------------------------------------------
// avg = mask @ x + fused branch-input select. Double-buffered LDS B staging.
// Block: 8 waves, tile 16 rows x 128 cols, K chunked by 32.
// ---------------------------------------------------------------------------
__global__ __launch_bounds__(256)
void avg_gemm(const unsigned short* __restrict__ maskb,
              const float* __restrict__ selscale,
              const float* __restrict__ xf,
              unsigned short* __restrict__ outb) {
    __shared__ unsigned short Bt[2][128 * 34];   // pitch 34: conflict-free
    const int colbase = blockIdx.x * 128;
    const int ibase   = blockIdx.y * 16;
    const int tid  = threadIdx.x;
    const int wave = tid >> 5;
    const int lane = tid & 31;
    const int lane16 = lane & 15;
    const int hi = lane >> 4;

    const unsigned short* Arow = maskb + (size_t)(ibase + lane16) * 512;
    v8f acc = {};

    auto stage = [&](int chunk, int buf) {
        const int j0 = chunk * 32;
        #pragma unroll
        for (int t = 0; t < 16; ++t) {               // 128*32/256
            int i  = tid + t * 256;
            int kk = i >> 7, p = i & 127;
            float f = xf[(size_t)(j0 + kk) * COLS + colbase + p];  // coalesced
            Bt[buf][p * 34 + kk] = f2bf(f);
        }
    };

    stage(0, 0);
    for (int chunk = 0; chunk < 16; ++chunk) {
        __syncthreads();
        if (chunk + 1 < 16) stage(chunk + 1, (chunk + 1) & 1);
        const unsigned short* B = Bt[chunk & 1];
        ABfrag a, b;
        load_a_frag(a, Arow + chunk * 32 + hi * 16);
        const unsigned short* bp = B + (wave * 16 + lane16) * 34 + hi * 16;
        #pragma unroll
        for (int q = 0; q < 8; ++q)
            b.u[q] = *(const unsigned int*)(bp + q * 2);
        acc = __builtin_amdgcn_wmma_f32_16x16x32_bf16(
            false, a.v, false, b.v, (short)0, acc, false, false);
    }

    const int col = colbase + wave * 16 + lane16;
    #pragma unroll
    for (int r = 0; r < 8; ++r) {
        int i = ibase + r + hi * 8;
        float ss = selscale[i];
        float xv = xf[(size_t)i * COLS + col];
        float v = (ss != 0.f) ? acc[r] * ss : xv;    // no fdiv, branchless
        outb[(size_t)i * COLS + col] = f2bf(v);
    }
}

// ---------------------------------------------------------------------------
// 3x3 implicit-GEMM conv (pad 1), bf16 WMMA, double-buffered LDS im2col.
// Block: one image, 128 out-ch x 64 px; wave = 16 ch x 4 pixel-subtiles.
// ---------------------------------------------------------------------------
template <int MODE, bool PERIMG>
__global__ __launch_bounds__(256)
void conv3_wmma(const unsigned short* __restrict__ in,
                const unsigned short* __restrict__ wpack,
                const int* __restrict__ classes,
                const float* __restrict__ xf,  const float* __restrict__ gcf,
                const float* __restrict__ bnS, const float* __restrict__ bnB,
                unsigned short* __restrict__ out_b, float* __restrict__ out_f) {
    constexpr int KDIM = 2304, NCHUNK = 72;
    __shared__ unsigned short Bt[2][64 * 34];

    const int n = blockIdx.z;
    const int mbase = blockIdx.y * 128;
    const int colbase = blockIdx.x * 64;
    const int tid = threadIdx.x;
    const int wave = tid >> 5;
    const int lane = tid & 31;
    const int lane16 = lane & 15;
    const int hi = lane >> 4;

    const unsigned short* Wbase = wpack;
    if constexpr (PERIMG) {
        if (classes[n] != 0) Wbase += (size_t)256 * KDIM;   // sel 1 = char
    }
    const unsigned short* Arow = Wbase + (size_t)(mbase + wave * 16 + lane16) * KDIM;
    const unsigned short* inn = in + (size_t)n * FEAT_C * FEAT_HW;

    v8f acc[4] = {};

    auto stage = [&](int chunk, int buf) {
        int kpos = chunk >> 3;
        int c0 = (chunk & 7) * 32;
        int dy = kpos / 3 - 1, dx = kpos % 3 - 1;
        #pragma unroll
        for (int t = 0; t < 8; ++t) {                // 64*32/256
            int i  = tid + t * 256;
            int kk = i >> 6, p = i & 63;
            int pixel = colbase + p;
            unsigned short v = 0;
            if (pixel < FEAT_HW) {
                int py = pixel / 14 + dy, px = pixel % 14 + dx;
                if (py >= 0 && py < 14 && px >= 0 && px < 14)
                    v = inn[(size_t)(c0 + kk) * FEAT_HW + py * 14 + px];
            }
            Bt[buf][p * 34 + kk] = v;
        }
    };

    stage(0, 0);
    for (int chunk = 0; chunk < NCHUNK; ++chunk) {
        __syncthreads();
        if (chunk + 1 < NCHUNK) stage(chunk + 1, (chunk + 1) & 1);  // overlap
        const unsigned short* B = Bt[chunk & 1];

        ABfrag a;
        load_a_frag(a, Arow + chunk * 32 + hi * 16);   // 2x global_load_b128
        if (chunk + 1 < NCHUNK)
            __builtin_prefetch(Arow + (chunk + 1) * 32, 0, 0);

        #pragma unroll
        for (int s = 0; s < 4; ++s) {
            ABfrag b;
            const unsigned short* bp = B + (s * 16 + lane16) * 34 + hi * 16;
            #pragma unroll
            for (int q = 0; q < 8; ++q)
                b.u[q] = *(const unsigned int*)(bp + q * 2);
            acc[s] = __builtin_amdgcn_wmma_f32_16x16x32_bf16(
                false, a.v, false, b.v, (short)0, acc[s], false, false);
        }
    }
    conv_epilogue<MODE>(acc, n, mbase, wave, lane16, hi, colbase,
                        xf, gcf, bnS, bnB, out_b, out_f);
}

// ---------------------------------------------------------------------------
// 1x1 conv with Tensor Data Mover staging: each K-chunk's B tile is a clean
// 2D tile (32 ch x 64 px, row stride 196) of the bf16 input tensor — one
// tensor_load_to_lds per chunk (wave 0 issues, s_wait_tensorcnt + barrier),
// double-buffered. LDS tile layout is [kk][64 px] (TDM copies tensor order).
// TDM OOB-reads-return-zero covers the partial last pixel group.
// ---------------------------------------------------------------------------
template <int MODE, bool PERIMG>
__global__ __launch_bounds__(256)
void conv1_wmma_tdm(const unsigned short* __restrict__ in,
                    const unsigned short* __restrict__ wpack,
                    const int* __restrict__ classes,
                    const float* __restrict__ xf,  const float* __restrict__ gcf,
                    const float* __restrict__ bnS, const float* __restrict__ bnB,
                    unsigned short* __restrict__ out_b, float* __restrict__ out_f) {
    constexpr int KDIM = 256, NCHUNK = 8;
    __shared__ unsigned short BtT[2][32 * 64];      // [kk][pixel]

    const int n = blockIdx.z;
    const int mbase = blockIdx.y * 128;
    const int colbase = blockIdx.x * 64;
    const int tid = threadIdx.x;
    const int wave = tid >> 5;
    const int lane = tid & 31;
    const int lane16 = lane & 15;
    const int hi = lane >> 4;

    const unsigned short* Wbase = wpack;
    if constexpr (PERIMG) {
        if (classes[n] != 0) Wbase += (size_t)256 * KDIM;
    }
    const unsigned short* Arow = Wbase + (size_t)(mbase + wave * 16 + lane16) * KDIM;
    const unsigned short* inn = in + (size_t)n * FEAT_C * FEAT_HW;

    v8f acc[4] = {};

#if HAVE_TDM
    auto tdm_issue = [&](int chunk, int buf) {
        const int c0 = chunk * 32;
        unsigned long long ga = (unsigned long long)(const void*)inn
                              + ((size_t)c0 * FEAT_HW + colbase) * 2ull;
        unsigned lds_off = (unsigned)(unsigned long long)(void*)&BtT[buf][0];
        // D# group 0 (ISA 8.3): count=1 | lds_addr | global_addr | type=2
        v4u g0;
        g0.x = 1u;
        g0.y = lds_off;
        g0.z = (unsigned)(ga & 0xFFFFFFFFull);
        g0.w = (unsigned)((ga >> 32) & 0x01FFFFFFull) | (2u << 30);
        // D# group 1 (ISA 8.4): data_size=1(2B); tensor_dim0/1 from tile
        // start (OOB x/y -> zero fill); tile 64x32; dim0 stride 196.
        v8i g1;
        g1[0] = (1 << 16);                           // data_size = 2 bytes
        g1[1] = (int)((unsigned)(FEAT_HW - colbase) << 16);   // tensor_dim0 lo16
        g1[2] = (int)((unsigned)(FEAT_C - c0) << 16);         // dim0 hi | tensor_dim1 lo16
        g1[3] = (int)(64u << 16);                    // dim1 hi | tile_dim0 = 64
        g1[4] = 32;                                  // tile_dim1 = 32, tile_dim2 = 0
        g1[5] = FEAT_HW;                             // tensor_dim0_stride lo32
        g1[6] = 0;
        g1[7] = 0;
        v4i zz4 = {0, 0, 0, 0};
        v8i zz8 = {0, 0, 0, 0, 0, 0, 0, 0};
        // 6-arg form (clang-23 / therock headers): g0, g1, g2, g3, g4, cpol
        __builtin_amdgcn_tensor_load_to_lds(g0, g1, zz4, zz4, zz8, 0);
    };
    if (wave == 0) {
        tdm_issue(0, 0);
        __builtin_amdgcn_s_wait_tensorcnt(0);
    }
#else
    auto stage = [&](int chunk, int buf) {
        const int c0 = chunk * 32;
        #pragma unroll
        for (int t = 0; t < 8; ++t) {
            int i  = tid + t * 256;
            int kk = i >> 6, p = i & 63;
            int pixel = colbase + p;
            BtT[buf][kk * 64 + p] =
                (pixel < FEAT_HW) ? inn[(size_t)(c0 + kk) * FEAT_HW + pixel] : 0;
        }
    };
    stage(0, 0);
#endif

    for (int chunk = 0; chunk < NCHUNK; ++chunk) {
        __syncthreads();
#if HAVE_TDM
        if (wave == 0 && chunk + 1 < NCHUNK) tdm_issue(chunk + 1, (chunk + 1) & 1);
#else
        if (chunk + 1 < NCHUNK) stage(chunk + 1, (chunk + 1) & 1);
#endif
        const unsigned short* B = BtT[chunk & 1];

        ABfrag a;
        load_a_frag(a, Arow + chunk * 32 + hi * 16);

        #pragma unroll
        for (int s = 0; s < 4; ++s) {
            ABfrag b;
            int p = s * 16 + lane16;
            #pragma unroll
            for (int e = 0; e < 16; ++e)             // K = hi*16 + e, column p
                b.h[e] = B[(hi * 16 + e) * 64 + p];
            acc[s] = __builtin_amdgcn_wmma_f32_16x16x32_bf16(
                false, a.v, false, b.v, (short)0, acc[s], false, false);
        }
#if HAVE_TDM
        if (wave == 0 && chunk + 1 < NCHUNK)
            __builtin_amdgcn_s_wait_tensorcnt(0);    // next tile landed
#endif
    }
    conv_epilogue<MODE>(acc, n, mbase, wave, lane16, hi, colbase,
                        xf, gcf, bnS, bnB, out_b, out_f);
}

// ---------------------------------------------------------------------------
extern "C" void kernel_launch(void* const* d_in, const int* in_sizes, int n_in,
                              void* d_out, int out_size, void* d_ws, size_t ws_size,
                              hipStream_t stream) {
    const float* x      = (const float*)d_in[0];
    const float* gc     = (const float*)d_in[1];
    const float* boxes  = (const float*)d_in[2];
    const int*   classes= (const int*)  d_in[3];
    const float* Wc3    = (const float*)d_in[4];
    const float* Wc1    = (const float*)d_in[5];
    const float* Wt3    = (const float*)d_in[6];
    const float* Wt1    = (const float*)d_in[7];
    const float* Wf3    = (const float*)d_in[8];
    const float* Wf1    = (const float*)d_in[9];
    const float* gamma  = (const float*)d_in[10];
    const float* beta   = (const float*)d_in[11];
    const float* mean   = (const float*)d_in[12];
    const float* var    = (const float*)d_in[13];

    char* ws = (char*)d_ws;
    const size_t FEAT = (size_t)FEAT_N * FEAT_C * FEAT_HW;
    auto au = [](size_t v) { return (v + 255) & ~(size_t)255; };
    size_t off = 0;
    unsigned short* buf0  = (unsigned short*)(ws + off); off = au(off + FEAT * 2);
    unsigned short* buf1  = (unsigned short*)(ws + off); off = au(off + FEAT * 2);
    unsigned short* maskb = (unsigned short*)(ws + off); off = au(off + (size_t)512 * 512 * 2);
    float*          selsc = (float*)         (ws + off); off = au(off + 512 * 4);
    unsigned short* Wb3   = (unsigned short*)(ws + off); off = au(off + (size_t)2 * 256 * 2304 * 2);
    unsigned short* Wb1   = (unsigned short*)(ws + off); off = au(off + (size_t)2 * 256 * 256 * 2);
    unsigned short* Wf3p  = (unsigned short*)(ws + off); off = au(off + (size_t)256 * 2304 * 2);
    unsigned short* Wf1p  = (unsigned short*)(ws + off); off = au(off + (size_t)256 * 256 * 2);
    float*          bnS   = (float*)         (ws + off); off = au(off + 256 * 4);
    float*          bnB   = (float*)         (ws + off); off = au(off + 256 * 4);
    // ~107 MB total

    // 1) pack weights bf16 K-major (sel0 = text, sel1 = char); fold BN
    pack_w<<<dim3((256 * 2304 + 255) / 256), 256, 0, stream>>>(Wt3, Wb3, 9);
    pack_w<<<dim3((256 * 2304 + 255) / 256), 256, 0, stream>>>(Wc3, Wb3 + (size_t)256 * 2304, 9);
    pack_w<<<dim3((256 * 256 + 255) / 256), 256, 0, stream>>>(Wt1, Wb1, 1);
    pack_w<<<dim3((256 * 256 + 255) / 256), 256, 0, stream>>>(Wc1, Wb1 + (size_t)256 * 256, 1);
    pack_w<<<dim3((256 * 2304 + 255) / 256), 256, 0, stream>>>(Wf3, Wf3p, 9);
    pack_w<<<dim3((256 * 256 + 255) / 256), 256, 0, stream>>>(Wf1, Wf1p, 1);
    bn_prep<<<1, 256, 0, stream>>>(gamma, beta, mean, var, bnS, bnB);

    // 2) pairwise mask + fused class/count scale
    mask_kernel<<<512, 512, 0, stream>>>(boxes, classes, maskb, selsc);

    // 3) gather-average GEMM + branch-input select -> buf0 (bf16)
    avg_gemm<<<dim3(COLS / 128, 512 / 16, 1), 256, 0, stream>>>(maskb, selsc, x, buf0);

    // 4) branch conv3x3 (per-image weights) -> buf1
    conv3_wmma<0, true><<<dim3(4, 2, 512), 256, 0, stream>>>(
        buf0, Wb3, classes, nullptr, nullptr, nullptr, nullptr, buf1, nullptr);
    // 5) branch conv1x1 (per-image, TDM staging) + x + gc -> buf0
    conv1_wmma_tdm<1, true><<<dim3(4, 2, 512), 256, 0, stream>>>(
        buf1, Wb1, classes, x, gc, nullptr, nullptr, buf0, nullptr);
    // 6) fuse conv3x3 -> buf1
    conv3_wmma<0, false><<<dim3(4, 2, 512), 256, 0, stream>>>(
        buf0, Wf3p, nullptr, nullptr, nullptr, nullptr, nullptr, buf1, nullptr);
    // 7) fuse conv1x1 (TDM staging) + BN + ReLU -> d_out (fp32)
    conv1_wmma_tdm<2, false><<<dim3(4, 2, 512), 256, 0, stream>>>(
        buf1, Wf1p, nullptr, nullptr, nullptr, bnS, bnB, nullptr, (float*)d_out);
}
